// MultiLayerGCN_variate_87325275062333
// MI455X (gfx1250) — matrix operand order repaired and checked by reading, not compiled, not run on hardware
//
#include <hip/hip_runtime.h>

// ---- problem dims ----
#define B_   32
#define M_   512
#define L_   512
#define D_   128
#define H_   8
#define HD_  16
#define DFF_ 256
#define K_   16
#define NL_  2
#define N_   (B_*M_)

typedef __attribute__((ext_vector_type(2)))  float        v2f;
typedef __attribute__((ext_vector_type(8)))  float        v8f;
typedef __attribute__((ext_vector_type(16))) __bf16       v16bf;
typedef __attribute__((ext_vector_type(8)))  unsigned int v8u;

__device__ __forceinline__ unsigned short f2bf(float f) {
    unsigned int u = __builtin_bit_cast(unsigned int, f);
    unsigned int r = u + 0x7fffu + ((u >> 16) & 1u);
    return (unsigned short)(r >> 16);
}

// -------- elementwise helpers --------
__global__ void k_f2bf(const float* __restrict__ in, unsigned short* __restrict__ out, int n) {
    int i = blockIdx.x * blockDim.x + threadIdx.x;
    if (i < n) out[i] = f2bf(in[i]);
}
// convert + transpose each [Kd x Nd] matrix (batched) -> bf16 [Nd x Kd]
__global__ void k_f2bf_t(const float* __restrict__ in, unsigned short* __restrict__ out,
                         int Kd, int Nd, int total) {
    int i = blockIdx.x * blockDim.x + threadIdx.x;
    if (i < total) {
        int mat = i / (Kd * Nd);
        int rem = i - mat * Kd * Nd;
        int k = rem / Nd, n = rem - k * Nd;
        out[(size_t)mat * Kd * Nd + (size_t)n * Kd + k] = f2bf(in[i]);
    }
}
__global__ void k_copy(const float* __restrict__ in, float* __restrict__ out, int n) {
    int i = blockIdx.x * blockDim.x + threadIdx.x;
    if (i < n) out[i] = in[i];
}
__global__ void k_fill(float* __restrict__ p, float v, int n) {
    int i = blockIdx.x * blockDim.x + threadIdx.x;
    if (i < n) p[i] = v;
}
__global__ void k_rsqrt_inplace(float* __restrict__ p, int n) {
    int i = blockIdx.x * blockDim.x + threadIdx.x;
    if (i < n) { float d = p[i]; p[i] = d > 0.f ? rsqrtf(d) : 0.f; }
}

// -------- center + unit-normalize rows of x (transposed view of x_enc) --------
// cn[b,m,l] = (x_enc[b,l,m] - mean_l) / ||.||_2  so corr = cn @ cn^T exactly.
__global__ void __launch_bounds__(128) k_center_norm(const float* __restrict__ xe,
                                                     float* __restrict__ cn) {
    int row = blockIdx.x;               // b*M + m
    int b = row / M_, m = row % M_;
    int t = threadIdx.x;
    __shared__ float red[128];
    float xr[L_ / 128];
    float s = 0.f;
#pragma unroll
    for (int j = 0; j < L_ / 128; ++j) {
        int l = t + j * 128;
        float x = xe[((size_t)b * L_ + l) * M_ + m];
        xr[j] = x; s += x;
    }
    red[t] = s; __syncthreads();
    for (int o = 64; o > 0; o >>= 1) { if (t < o) red[t] += red[t + o]; __syncthreads(); }
    float mean = red[0] / (float)L_;
    __syncthreads();
    float sq = 0.f;
#pragma unroll
    for (int j = 0; j < L_ / 128; ++j) { xr[j] -= mean; sq += xr[j] * xr[j]; }
    red[t] = sq; __syncthreads();
    for (int o = 64; o > 0; o >>= 1) { if (t < o) red[t] += red[t + o]; __syncthreads(); }
    float inv = red[0] > 0.f ? rsqrtf(red[0]) : 0.f;
#pragma unroll
    for (int j = 0; j < L_ / 128; ++j)
        cn[(size_t)row * L_ + t + j * 128] = xr[j] * inv;
}

// -------- corr = cn @ cn^T via fp32 WMMA 16x16x4 (one wave per 16x16 tile) --------
__global__ void __launch_bounds__(32) k_corr(const float* __restrict__ cn,
                                             float* __restrict__ corr) {
    int b = blockIdx.z;
    int m0 = blockIdx.x << 4, n0 = blockIdx.y << 4;
    int lane = threadIdx.x, ml = lane & 15, kg = lane >> 4;
    const float* am = cn + ((size_t)b * M_ + m0 + ml) * L_;
    const float* bn = cn + ((size_t)b * M_ + n0 + ml) * L_;
    v8f acc = {0.f, 0.f, 0.f, 0.f, 0.f, 0.f, 0.f, 0.f};
    for (int k0 = 0; k0 < L_; k0 += 4) {
        v2f a, bb;
        const float* ap = am + k0 + kg * 2;
        a.x = ap[0]; a.y = ap[1];
        const float* bp = bn + k0 + kg * 2;   // B[k][n] = cn[n][k]
        bb.x = bp[0]; bb.y = bp[1];
        acc = __builtin_amdgcn_wmma_f32_16x16x4_f32(false, a, false, bb,
                                                    (short)0, acc, false, false);
    }
#pragma unroll
    for (int v = 0; v < 8; ++v) {
        int r = m0 + v + (kg << 3);
        corr[((size_t)b * M_ + r) * M_ + n0 + ml] = acc[v];
    }
}

// -------- stable ascending top-(K+1) mins per row; keep ranks 1..K --------
__global__ void __launch_bounds__(32) k_topk(const float* __restrict__ corr,
                                             int* __restrict__ nbr) {
    int row = blockIdx.x, lane = threadIdx.x;
    const float* cr = corr + (size_t)row * M_;
    float vals[M_ / 32];
#pragma unroll
    for (int j = 0; j < M_ / 32; ++j) vals[j] = cr[lane + j * 32];
    for (int t = 0; t < K_ + 1; ++t) {
        float bv = 3.0e38f; int bc = 0x7fffffff;
#pragma unroll
        for (int j = 0; j < M_ / 32; ++j) {
            int c = lane + j * 32;
            if (vals[j] < bv || (vals[j] == bv && c < bc)) { bv = vals[j]; bc = c; }
        }
        for (int off = 1; off < 32; off <<= 1) {
            float ov = __shfl_xor(bv, off, 32);
            int   oc = __shfl_xor(bc, off, 32);
            if (ov < bv || (ov == bv && oc < bc)) { bv = ov; bc = oc; }
        }
        if (t > 0 && lane == 0) nbr[(size_t)row * K_ + (t - 1)] = bc;
        if ((bc & 31) == lane) vals[bc >> 5] = 3.4e38f;
    }
}

// -------- degree accumulation over dst occurrences --------
__global__ void k_deg_accum(const int* __restrict__ nbr, float* __restrict__ deg) {
    int i = blockIdx.x * blockDim.x + threadIdx.x;
    if (i < B_ * M_ * K_) {
        int b = i / (M_ * K_);
        atomicAdd(&deg[b * M_ + nbr[i]], 1.0f);
    }
}

// -------- bf16 GEMM: C = act(A[MxK] @ B[KxN] + bias), B pre-transposed [N x K] --------
// One wave computes a 16x64 tile (4 accumulators); A fragment + epilogue amortized 4x.
// With Bt transposed, both operands load contiguous 32B runs per lane (b128-mergeable).
__global__ void __launch_bounds__(32) k_gemm_bf16(
    const unsigned short* __restrict__ A, const unsigned short* __restrict__ Bt,
    const float* __restrict__ bias, float* __restrict__ Cf,
    unsigned short* __restrict__ Cb, int Mrows, int Ndim, int Kdim, int relu) {
    int lane = threadIdx.x, ml = lane & 15, kg = lane >> 4;
    int m0 = blockIdx.x << 4, n0 = blockIdx.y << 6;
    v8f acc[4];
#pragma unroll
    for (int j = 0; j < 4; ++j) acc[j] = (v8f){0.f,0.f,0.f,0.f,0.f,0.f,0.f,0.f};
    const unsigned int* arow = (const unsigned int*)(A + (size_t)(m0 + ml) * Kdim);
    const unsigned int* brow[4];
#pragma unroll
    for (int j = 0; j < 4; ++j)
        brow[j] = (const unsigned int*)(Bt + (size_t)(n0 + 16 * j + ml) * Kdim);
    for (int k0 = 0; k0 < Kdim; k0 += 32) {
        int kw0 = k0 >> 1;
        v8u aw;
#pragma unroll
        for (int v = 0; v < 8; ++v)        // A 16x32 bf16 layout (pair words)
            aw[v] = arow[kw0 + (kg << 3) + v];
        v16bf av = __builtin_bit_cast(v16bf, aw);
#pragma unroll
        for (int j = 0; j < 4; ++j) {
            v8u bw;
#pragma unroll
            for (int v = 0; v < 8; ++v)    // B 32x16: contiguous k-pairs from Bt row
                bw[v] = brow[j][kw0 + (kg << 3) + v];
            v16bf bv = __builtin_bit_cast(v16bf, bw);
            acc[j] = __builtin_amdgcn_wmma_f32_16x16x32_bf16(false, av, false, bv,
                                                             (short)0, acc[j], false, false);
        }
    }
#pragma unroll
    for (int j = 0; j < 4; ++j) {
#pragma unroll
        for (int v = 0; v < 8; ++v) {
            int r = m0 + v + (kg << 3), c = n0 + 16 * j + ml;
            float x = acc[j][v];
            if (bias) x += bias[c];
            if (relu) x = fmaxf(x, 0.f);
            size_t idx = (size_t)r * Ndim + c;
            if (Cf) Cf[idx] = x;
            if (Cb) Cb[idx] = f2bf(x);
        }
    }
}

// -------- GCN edge aggregation (faithful mismatched src pairing) --------
__global__ void __launch_bounds__(128) k_edge_agg(const float* __restrict__ h,
                                                  const int* __restrict__ nbr,
                                                  const float* __restrict__ dinv,
                                                  float* __restrict__ agg) {
    int e = blockIdx.x;
    int b = e / (M_ * K_);
    int r = e - b * (M_ * K_);
    int src = r % M_;                  // row = tile(arange(M), K)
    int dst = nbr[e];                  // col = neighbors.reshape(M*K)
    float nrm = dinv[b * M_ + src] * dinv[b * M_ + dst];
    int t = threadIdx.x;
    atomicAdd(&agg[((size_t)(b * M_ + dst)) * D_ + t],
              h[((size_t)(b * M_ + src)) * D_ + t] * nrm);
}

// -------- GCN epilogue: agg + self-loop + bias, ReLU --------
__global__ void __launch_bounds__(128) k_gcn_finish(const float* __restrict__ agg,
                                                    const float* __restrict__ hw,
                                                    const float* __restrict__ dinv,
                                                    const float* __restrict__ bias,
                                                    float* __restrict__ of,
                                                    unsigned short* __restrict__ ob) {
    int n = blockIdx.x, t = threadIdx.x;
    float di = dinv[n];
    float x = agg[(size_t)n * D_ + t] + hw[(size_t)n * D_ + t] * di * di + bias[t];
    x = fmaxf(x, 0.f);
    if (of) of[(size_t)n * D_ + t] = x;
    if (ob) ob[(size_t)n * D_ + t] = f2bf(x);
}

// -------- flash attention: one wave per (qtile, head, batch), hd=16 padded to K=32 --------
__global__ void __launch_bounds__(32) k_flash(const unsigned short* __restrict__ Q,
                                              const unsigned short* __restrict__ Kt,
                                              const unsigned short* __restrict__ Vt,
                                              unsigned short* __restrict__ Ot,
                                              float scale) {
    int qt = blockIdx.x, hh = blockIdx.y, b = blockIdx.z;
    int lane = threadIdx.x, ml = lane & 15, kg = lane >> 4;
    __shared__ unsigned short ptile[16][16];

    v8u qw;
    const unsigned int* qrow =
        (const unsigned int*)(Q + (((size_t)(b * M_ + qt * 16 + ml)) * H_ + hh) * HD_);
#pragma unroll
    for (int v = 0; v < 8; ++v) {
        int p = ((v & 4) << 2) + ((v & 3) << 1) + (kg << 3);
        qw[v] = (p < HD_) ? qrow[p >> 1] : 0u;
    }
    v16bf qa = __builtin_bit_cast(v16bf, qw);

    v8f o = {0.f, 0.f, 0.f, 0.f, 0.f, 0.f, 0.f, 0.f};
    float mr[8], lr[8];
#pragma unroll
    for (int v = 0; v < 8; ++v) { mr[v] = -3.0e38f; lr[v] = 0.f; }

    for (int kt = 0; kt < M_ / 16; ++kt) {
        v8u kw;
        const unsigned int* krow =
            (const unsigned int*)(Kt + (((size_t)(b * M_ + kt * 16 + ml)) * H_ + hh) * HD_);
#pragma unroll
        for (int v = 0; v < 8; ++v) {
            int kk = ((v & 4) << 2) + ((v & 3) << 1) + (kg << 3);
            kw[v] = (kk < HD_) ? krow[kk >> 1] : 0u;
        }
        v16bf kb = __builtin_bit_cast(v16bf, kw);
        v8f z = {0.f, 0.f, 0.f, 0.f, 0.f, 0.f, 0.f, 0.f};
        v8f s = __builtin_amdgcn_wmma_f32_16x16x32_bf16(false, qa, false, kb,
                                                        (short)0, z, false, false);
        float pv[8];
#pragma unroll
        for (int v = 0; v < 8; ++v) {
            float sv = s[v] * scale;
            float rm = sv;
            for (int off = 1; off < 16; off <<= 1) rm = fmaxf(rm, __shfl_xor(rm, off, 32));
            float nm = fmaxf(mr[v], rm);
            float alpha = __expf(mr[v] - nm);
            mr[v] = nm;
            float p = __expf(sv - nm);
            pv[v] = p;
            float ps = p;
            for (int off = 1; off < 16; off <<= 1) ps += __shfl_xor(ps, off, 32);
            lr[v] = lr[v] * alpha + ps;
            o[v] = o[v] * alpha;
        }
        __syncthreads();
#pragma unroll
        for (int v = 0; v < 8; ++v) ptile[v + (kg << 3)][ml] = f2bf(pv[v]);
        __syncthreads();
        v8u pw;
#pragma unroll
        for (int v = 0; v < 8; ++v) {
            int p = ((v & 4) << 2) + ((v & 3) << 1) + (kg << 3);
            pw[v] = (p < 16) ? *(const unsigned int*)&ptile[ml][p] : 0u;
        }
        v16bf pa = __builtin_bit_cast(v16bf, pw);
        v8u vw;
#pragma unroll
        for (int v = 0; v < 8; ++v) {
            int kk = ((v & 4) << 2) + ((v & 3) << 1) + (kg << 3);
            if (kk < 16) {
                unsigned int lo = Vt[(((size_t)(b * M_ + kt * 16 + kk)) * H_ + hh) * HD_ + ml];
                unsigned int hi = Vt[(((size_t)(b * M_ + kt * 16 + kk + 1)) * H_ + hh) * HD_ + ml];
                vw[v] = lo | (hi << 16);
            } else vw[v] = 0u;
        }
        v16bf vb = __builtin_bit_cast(v16bf, vw);
        o = __builtin_amdgcn_wmma_f32_16x16x32_bf16(false, pa, false, vb,
                                                    (short)0, o, false, false);
        __syncthreads();
    }
#pragma unroll
    for (int v = 0; v < 8; ++v) {
        int r = qt * 16 + v + (kg << 3);
        float x = o[v] / lr[v];
        Ot[(((size_t)(b * M_ + r)) * H_ + hh) * HD_ + ml] = f2bf(x);
    }
}

// -------- fused residual + LayerNorm (writes f32 and optional bf16) --------
__global__ void __launch_bounds__(128) k_add_ln(const float* __restrict__ x,
                                                const float* __restrict__ y,
                                                const float* __restrict__ g,
                                                const float* __restrict__ bta,
                                                float* __restrict__ of,
                                                unsigned short* __restrict__ ob) {
    int row = blockIdx.x, t = threadIdx.x;
    __shared__ float red[128];
    float v = x[(size_t)row * D_ + t] + y[(size_t)row * D_ + t];
    red[t] = v; __syncthreads();
    for (int o = 64; o > 0; o >>= 1) { if (t < o) red[t] += red[t + o]; __syncthreads(); }
    float mu = red[0] / (float)D_;
    __syncthreads();
    float c = v - mu;
    red[t] = c * c; __syncthreads();
    for (int o = 64; o > 0; o >>= 1) { if (t < o) red[t] += red[t + o]; __syncthreads(); }
    float var = red[0] / (float)D_;
    float r = c * rsqrtf(var + 1e-5f) * g[t] + bta[t];
    if (of) of[(size_t)row * D_ + t] = r;
    if (ob) ob[(size_t)row * D_ + t] = f2bf(r);
}

extern "C" void kernel_launch(void* const* d_in, const int* in_sizes, int n_in,
                              void* d_out, int out_size, void* d_ws, size_t ws_size,
                              hipStream_t stream) {
    const float* enc     = (const float*)d_in[0];
    const float* x_enc   = (const float*)d_in[1];
    const float* conv1_W = (const float*)d_in[2];
    const float* conv1_b = (const float*)d_in[3];
    const float* conv2_W = (const float*)d_in[4];
    const float* conv2_b = (const float*)d_in[5];
    const float* Wq  = (const float*)d_in[6];
    const float* Wk  = (const float*)d_in[7];
    const float* Wv  = (const float*)d_in[8];
    const float* Wo  = (const float*)d_in[9];
    const float* fW1 = (const float*)d_in[10];
    const float* fb1 = (const float*)d_in[11];
    const float* fW2 = (const float*)d_in[12];
    const float* fb2 = (const float*)d_in[13];
    const float* ln1g = (const float*)d_in[14];
    const float* ln1b = (const float*)d_in[15];
    const float* ln2g = (const float*)d_in[16];
    const float* ln2b = (const float*)d_in[17];

    char* ws = (char*)d_ws;
    const size_t CORR_BYTES = (size_t)B_ * M_ * M_ * 4;   // 32 MiB
    // region 0 (corr until topk; then reused for f32 activations)
    float* CORR = (float*)(ws + 0);
    float* Hbuf = (float*)(ws + 8388608);
    float* Obuf = (float*)(ws + 16777216);
    // region 1 (cnorm until corr; then reused for bf16 activations)
    float* CNORM = (float*)(ws + CORR_BYTES);
    unsigned short* QB   = (unsigned short*)(ws + CORR_BYTES + 0);
    unsigned short* KB   = (unsigned short*)(ws + CORR_BYTES + 4194304);
    unsigned short* VB   = (unsigned short*)(ws + CORR_BYTES + 8388608);
    unsigned short* ATTB = (unsigned short*)(ws + CORR_BYTES + 12582912);
    unsigned short* F1B  = (unsigned short*)(ws + CORR_BYTES + 16777216);
    // region 2: persistent
    size_t base2 = 2 * CORR_BYTES;
    int*   NBR  = (int*)(ws + base2);
    float* DEG  = (float*)(ws + base2 + 1048576);
    unsigned short* HB  = (unsigned short*)(ws + base2 + 1114112);
    unsigned short* XGB = (unsigned short*)(ws + base2 + 1114112 + 4194304);
    float* HW   = (float*)(ws + base2 + 1114112 + 8388608);
    float* AGG  = (float*)(ws + base2 + 1114112 + 16777216);
    unsigned short* WPOOL = (unsigned short*)(ws + base2 + 1114112 + 25165824);
    unsigned short* c1W = WPOOL;                        // all weights stored transposed [N x K]
    unsigned short* c2W = c1W + 16384;
    unsigned short* WqB = c2W + 16384;
    unsigned short* WkB = WqB + NL_ * D_ * D_;
    unsigned short* WvB = WkB + NL_ * D_ * D_;
    unsigned short* WoB = WvB + NL_ * D_ * D_;
    unsigned short* W1B = WoB + NL_ * D_ * D_;
    unsigned short* W2B = W1B + NL_ * D_ * DFF_;

#define CDIV(a, b) (((a) + (b) - 1) / (b))
    // weight conversions to transposed bf16
    k_f2bf_t<<<CDIV(D_*D_,256),256,0,stream>>>(conv1_W, c1W, D_, D_, D_*D_);
    k_f2bf_t<<<CDIV(D_*D_,256),256,0,stream>>>(conv2_W, c2W, D_, D_, D_*D_);
    k_f2bf_t<<<CDIV(NL_*D_*D_,256),256,0,stream>>>(Wq, WqB, D_, D_, NL_*D_*D_);
    k_f2bf_t<<<CDIV(NL_*D_*D_,256),256,0,stream>>>(Wk, WkB, D_, D_, NL_*D_*D_);
    k_f2bf_t<<<CDIV(NL_*D_*D_,256),256,0,stream>>>(Wv, WvB, D_, D_, NL_*D_*D_);
    k_f2bf_t<<<CDIV(NL_*D_*D_,256),256,0,stream>>>(Wo, WoB, D_, D_, NL_*D_*D_);
    k_f2bf_t<<<CDIV(NL_*D_*DFF_,256),256,0,stream>>>(fW1, W1B, D_, DFF_, NL_*D_*DFF_);
    k_f2bf_t<<<CDIV(NL_*DFF_*D_,256),256,0,stream>>>(fW2, W2B, DFF_, D_, NL_*DFF_*D_);

    // graph construction
    k_center_norm<<<N_, 128, 0, stream>>>(x_enc, CNORM);
    k_corr<<<dim3(M_/16, M_/16, B_), 32, 0, stream>>>(CNORM, CORR);
    k_topk<<<N_, 32, 0, stream>>>(CORR, NBR);
    k_fill<<<CDIV(N_,256),256,0,stream>>>(DEG, 1.0f, N_);                   // self loops
    k_deg_accum<<<CDIV(B_*M_*K_,256),256,0,stream>>>(NBR, DEG);
    k_rsqrt_inplace<<<CDIV(N_,256),256,0,stream>>>(DEG, N_);

    // GCN
    k_f2bf<<<CDIV(N_*D_,256),256,0,stream>>>(enc, HB, N_*D_);
    k_gemm_bf16<<<dim3(N_/16, D_/64), 32, 0, stream>>>(HB, c1W, nullptr, HW, nullptr, N_, D_, D_, 0);
    k_fill<<<CDIV(N_*D_,256),256,0,stream>>>(AGG, 0.f, N_*D_);
    k_edge_agg<<<B_*M_*K_, 128, 0, stream>>>(HW, NBR, DEG, AGG);
    k_gcn_finish<<<N_, D_, 0, stream>>>(AGG, HW, DEG, conv1_b, nullptr, XGB);
    k_gemm_bf16<<<dim3(N_/16, D_/64), 32, 0, stream>>>(XGB, c2W, nullptr, HW, nullptr, N_, D_, D_, 0);
    k_fill<<<CDIV(N_*D_,256),256,0,stream>>>(AGG, 0.f, N_*D_);
    k_edge_agg<<<B_*M_*K_, 128, 0, stream>>>(HW, NBR, DEG, AGG);
    k_gcn_finish<<<N_, D_, 0, stream>>>(AGG, HW, DEG, conv2_b, nullptr, XGB);  // XGB = bf16(xg)

    // transformer (h starts as enc; HB already holds bf16(enc))
    k_copy<<<CDIV(N_*D_,256),256,0,stream>>>(enc, Hbuf, N_*D_);
    for (int l = 0; l < NL_; ++l) {
        const unsigned short* wq = WqB + (size_t)l * D_ * D_;
        const unsigned short* wk = WkB + (size_t)l * D_ * D_;
        const unsigned short* wv = WvB + (size_t)l * D_ * D_;
        const unsigned short* wo = WoB + (size_t)l * D_ * D_;
        const unsigned short* w1 = W1B + (size_t)l * D_ * DFF_;
        const unsigned short* w2 = W2B + (size_t)l * DFF_ * D_;
        k_gemm_bf16<<<dim3(N_/16, D_/64), 32, 0, stream>>>(HB,  wq, nullptr, nullptr, QB, N_, D_, D_, 0);
        k_gemm_bf16<<<dim3(N_/16, D_/64), 32, 0, stream>>>(XGB, wk, nullptr, nullptr, KB, N_, D_, D_, 0);
        k_gemm_bf16<<<dim3(N_/16, D_/64), 32, 0, stream>>>(XGB, wv, nullptr, nullptr, VB, N_, D_, D_, 0);
        k_flash<<<dim3(M_/16, H_, B_), 32, 0, stream>>>(QB, KB, VB, ATTB, 0.25f);
        k_gemm_bf16<<<dim3(N_/16, D_/64), 32, 0, stream>>>(ATTB, wo, nullptr, Obuf, nullptr, N_, D_, D_, 0);
        k_add_ln<<<N_, D_, 0, stream>>>(Hbuf, Obuf, ln1g + l*D_, ln1b + l*D_, Hbuf, HB);
        k_gemm_bf16<<<dim3(N_/16, DFF_/64), 32, 0, stream>>>(HB, w1, fb1 + l*DFF_, nullptr, F1B, N_, DFF_, D_, 1);
        k_gemm_bf16<<<dim3(N_/16, D_/64), 32, 0, stream>>>(F1B, w2, fb2 + l*D_, Obuf, nullptr, N_, D_, DFF_, 0);
        float* outf = (l == NL_ - 1) ? (float*)d_out : Hbuf;
        unsigned short* outb = (l == NL_ - 1) ? nullptr : HB;
        k_add_ln<<<N_, D_, 0, stream>>>(Hbuf, Obuf, ln2g + l*D_, ln2b + l*D_, outf, outb);
    }
#undef CDIV
    (void)in_sizes; (void)n_in; (void)out_size; (void)ws_size;
}
